// GNN_2345052143970
// MI455X (gfx1250) — compile-verified, matrix operand order checked by reading, not run on hardware
//
#include <hip/hip_runtime.h>
#include <hip/hip_bf16.h>
#include <math.h>

// ---------------------------------------------------------------------------
// Types for CDNA5 WMMA (wave32): v_wmma_f32_16x16x32_bf16
// ---------------------------------------------------------------------------
typedef __bf16 bf16_t;
typedef bf16_t v16bf __attribute__((ext_vector_type(16)));
typedef float  v8f   __attribute__((ext_vector_type(8)));

__device__ __forceinline__ unsigned short f2bf(float f) {
  unsigned int u = __float_as_uint(f);
  u += 0x7FFFu + ((u >> 16) & 1u);           // round-to-nearest-even
  return (unsigned short)(u >> 16);
}

__device__ __forceinline__ void store_bf4(unsigned short* p, float a, float b, float c, float d) {
  ushort4 v;
  v.x = f2bf(a); v.y = f2bf(b); v.z = f2bf(c); v.w = f2bf(d);
  *(ushort4*)p = v;
}

static inline int ceil_div(int a, int b) { return (a + b - 1) / b; }

// ---------------------------------------------------------------------------
// Elementwise helper kernels (float4 vectorized; all widths are multiples of 4)
// ---------------------------------------------------------------------------
__global__ void k_fill_zero4(float4* __restrict__ p, int n4) {
  int i = blockIdx.x * blockDim.x + threadIdx.x;
  if (i < n4) p[i] = make_float4(0.f, 0.f, 0.f, 0.f);
}

__global__ void k_copy_convert4(const float4* __restrict__ in, float4* __restrict__ xf,
                                unsigned short* __restrict__ xb, int n4) {
  int i = blockIdx.x * blockDim.x + threadIdx.x;
  if (i >= n4) return;
  float4 v = in[i];
  xf[i] = v;
  store_bf4(xb + 4 * i, v.x, v.y, v.z, v.w);
}

__global__ void k_to_bf16_4(const float4* __restrict__ in, unsigned short* __restrict__ out, int n4) {
  int i = blockIdx.x * blockDim.x + threadIdx.x;
  if (i >= n4) return;
  float4 v = in[i];
  store_bf4(out + 4 * i, v.x, v.y, v.z, v.w);
}

__device__ __forceinline__ float elu1(float v) { return v > 0.f ? v : expm1f(v); }

// ELU (+ optional eval-BatchNorm) + zero-pad Cin -> Cout; writes f32 + bf16 copies
__global__ void k_transition(const float* __restrict__ h, int Cin,
                             float* __restrict__ xf, unsigned short* __restrict__ xb, int Cout,
                             const float* __restrict__ gamma, const float* __restrict__ beta,
                             const float* __restrict__ mean, const float* __restrict__ var,
                             int N) {
  int cq = Cout >> 2;
  int i = blockIdx.x * blockDim.x + threadIdx.x;
  if (i >= N * cq) return;
  int n = i / cq, c4 = (i - n * cq) << 2;
  float4 v = make_float4(0.f, 0.f, 0.f, 0.f);
  if (c4 < Cin) {                                  // Cin multiple of 4: quad fully in or out
    v = *(const float4*)(h + (size_t)n * Cin + c4);
    v.x = elu1(v.x); v.y = elu1(v.y); v.z = elu1(v.z); v.w = elu1(v.w);
    if (gamma) {
      float4 g = *(const float4*)(gamma + c4);
      float4 b = *(const float4*)(beta + c4);
      float4 m = *(const float4*)(mean + c4);
      float4 s = *(const float4*)(var + c4);
      v.x = (v.x - m.x) * rsqrtf(s.x + 1e-5f) * g.x + b.x;
      v.y = (v.y - m.y) * rsqrtf(s.y + 1e-5f) * g.y + b.y;
      v.z = (v.z - m.z) * rsqrtf(s.z + 1e-5f) * g.z + b.z;
      v.w = (v.w - m.w) * rsqrtf(s.w + 1e-5f) * g.w + b.w;
    }
  }
  *(float4*)(xf + (size_t)n * Cout + c4) = v;
  store_bf4(xb + (size_t)n * Cout + c4, v.x, v.y, v.z, v.w);
}

// agg[dst] += m[src] over all edges (float4 gather, 4x f32 atomics)
__global__ void k_scatter_add(const float* __restrict__ m, const long long* __restrict__ ei,
                              float* __restrict__ agg, int E, int C) {
  int cq = C >> 2;
  int i = blockIdx.x * blockDim.x + threadIdx.x;
  if (i >= E * cq) return;
  int e  = i / cq;
  int c4 = (i - e * cq) << 2;
  int src = (int)ei[e];
  int dst = (int)ei[E + e];
  const float4 v = *(const float4*)(m + (size_t)src * C + c4);
  float* a = agg + (size_t)dst * C + c4;
  atomicAdd(a + 0, v.x); atomicAdd(a + 1, v.y);
  atomicAdd(a + 2, v.z); atomicAdd(a + 3, v.w);
}

__device__ __forceinline__ float sigm(float x) { return 1.f / (1.f + expf(-x)); }

// GRUCell gates; updates x (f32) and its bf16 shadow in place (4 channels/thread)
__global__ void k_gru(const float* __restrict__ gi, const float* __restrict__ gh,
                      float* __restrict__ xf, unsigned short* __restrict__ xb, int N, int C) {
  int cq = C >> 2;
  int i = blockIdx.x * blockDim.x + threadIdx.x;
  if (i >= N * cq) return;
  int n = i / cq, c4 = (i - n * cq) << 2;
  const float* gin = gi + (size_t)n * 3 * C + c4;
  const float* ghn = gh + (size_t)n * 3 * C + c4;
  float4 ir = *(const float4*)(gin);
  float4 iz = *(const float4*)(gin + C);
  float4 in_ = *(const float4*)(gin + 2 * C);
  float4 hr = *(const float4*)(ghn);
  float4 hz = *(const float4*)(ghn + C);
  float4 hn = *(const float4*)(ghn + 2 * C);
  float4 xo = *(const float4*)(xf + (size_t)n * C + c4);
  float4 xn;
  {
    float r = sigm(ir.x + hr.x), z = sigm(iz.x + hz.x), nn = tanhf(in_.x + r * hn.x);
    xn.x = (1.f - z) * nn + z * xo.x;
  }
  {
    float r = sigm(ir.y + hr.y), z = sigm(iz.y + hz.y), nn = tanhf(in_.y + r * hn.y);
    xn.y = (1.f - z) * nn + z * xo.y;
  }
  {
    float r = sigm(ir.z + hr.z), z = sigm(iz.z + hz.z), nn = tanhf(in_.z + r * hn.z);
    xn.z = (1.f - z) * nn + z * xo.z;
  }
  {
    float r = sigm(ir.w + hr.w), z = sigm(iz.w + hz.w), nn = tanhf(in_.w + r * hn.w);
    xn.w = (1.f - z) * nn + z * xo.w;
  }
  *(float4*)(xf + (size_t)n * C + c4) = xn;
  store_bf4(xb + (size_t)n * C + c4, xn.x, xn.y, xn.z, xn.w);
}

// global_add_pool (float4 gather, 4x f32 atomics)
__global__ void k_pool(const float* __restrict__ h, const long long* __restrict__ batch,
                       float* __restrict__ g, int N, int C) {
  int cq = C >> 2;
  int i = blockIdx.x * blockDim.x + threadIdx.x;
  if (i >= N * cq) return;
  int n = i / cq, c4 = (i - n * cq) << 2;
  float4 v = *(const float4*)(h + (size_t)n * C + c4);
  float* p = g + (size_t)batch[n] * C + c4;
  atomicAdd(p + 0, v.x); atomicAdd(p + 1, v.y);
  atomicAdd(p + 2, v.z); atomicAdd(p + 3, v.w);
}

// Pack a K x Ncols (or transposed Ncols x K) f32 weight into bf16 WMMA B-fragment
// order: dst[((nt*KCH + kk)*32 + lane)*16 + e], lane holds column nt*16+(lane&15),
// kbase = (lane>>4)*8 + kk*32, element e -> k = kbase + e + (e>=8 ? 8 : 0).
__global__ void k_pack_B(const float* __restrict__ src, unsigned short* __restrict__ dst,
                         int K, int Ncols, int trans) {
  int i = blockIdx.x * blockDim.x + threadIdx.x;
  if (i >= K * Ncols) return;
  int e    = i & 15;
  int lane = (i >> 4) & 31;
  int tile = i >> 9;
  int kch  = K >> 5;
  int nt   = tile / kch, kk = tile - nt * kch;
  int k = (kk << 5) + ((lane >> 4) << 3) + e + ((e >= 8) ? 8 : 0);
  int n = (nt << 4) + (lane & 15);
  float v = trans ? src[(size_t)n * K + k] : src[(size_t)k * Ncols + n];
  dst[i] = f2bf(v);
}

// ---------------------------------------------------------------------------
// WMMA GEMM: C[nrows,NC] = A[nrows,K](bf16) @ Bpacked + bias
// One wave per 16-row strip; A fragments live in VGPRs, reused across all
// N-tiles. Fully templated shape -> immediate-offset loads/stores, no branches.
// ---------------------------------------------------------------------------
template <int KCH, int NT, bool HB>
__global__ __launch_bounds__(256)
void k_wmma_gemm(const unsigned short* __restrict__ A,
                 const unsigned short* __restrict__ Bp,
                 const float* __restrict__ bias,
                 float* __restrict__ C, int nrows) {
  constexpr int K  = KCH * 32;
  constexpr int NC = NT * 16;
  const int lane  = threadIdx.x & 31;
  const int strip = (blockIdx.x * blockDim.x + threadIdx.x) >> 5;
  if (strip >= (nrows >> 4)) return;              // uniform per wave -> EXEC all-1 at WMMA
  const int l15 = lane & 15;
  const int hi8 = (lane >> 4) << 3;

  v16bf afrag[KCH];
  const unsigned short* ap = A + (size_t)((strip << 4) + l15) * K + hi8;
#pragma unroll
  for (int kk = 0; kk < KCH; ++kk) {
    union { uint4 u[2]; v16bf v; } ua;
    ua.u[0] = *(const uint4*)(ap + (kk << 5));        // K = kbase .. kbase+7
    ua.u[1] = *(const uint4*)(ap + (kk << 5) + 16);   // K = kbase+16 .. kbase+23
    afrag[kk] = ua.v;
  }

  const unsigned short* bp0 = Bp + (lane << 4);
  float* cp = C + (size_t)((strip << 4) + hi8) * NC + l15;
#pragma unroll
  for (int nt = 0; nt < NT; ++nt) {
    float bv = HB ? bias[(nt << 4) + l15] : 0.f;
    v8f acc = {bv, bv, bv, bv, bv, bv, bv, bv};
#pragma unroll
    for (int kk = 0; kk < KCH; ++kk) {
      union { uint4 u[2]; v16bf v; } ub;
      const unsigned short* bp = bp0 + (nt * KCH + kk) * 512;
      ub.u[0] = *(const uint4*)(bp);
      ub.u[1] = *(const uint4*)(bp + 8);
      acc = __builtin_amdgcn_wmma_f32_16x16x32_bf16(false, afrag[kk], false, ub.v,
                                                    (short)0, acc, false, false);
    }
#pragma unroll
    for (int r = 0; r < 8; ++r) cp[nt * 16 + r * NC] = acc[r];   // immediate offsets
  }
}

static void gemm(const unsigned short* A, const unsigned short* Bp, const float* bias,
                 float* Cmat, int nrows, int K, int Ncols, hipStream_t s) {
  int nstrips = nrows >> 4;
  dim3 blk(256);                                   // 8 waves/block
  dim3 grd(ceil_div(nstrips, 8));
#define GEMM_CASE(KC, NTV)                                                          \
  if (K == (KC)*32 && Ncols == (NTV)*16) {                                          \
    if (bias) k_wmma_gemm<KC, NTV, true><<<grd, blk, 0, s>>>(A, Bp, bias, Cmat, nrows); \
    else      k_wmma_gemm<KC, NTV, false><<<grd, blk, 0, s>>>(A, Bp, bias, Cmat, nrows); \
    return;                                                                         \
  }
  GEMM_CASE(1, 2)    // C=32:  x@W
  GEMM_CASE(1, 6)    // C=32:  GRU gates (3C=96)
  GEMM_CASE(2, 4)    // C=64:  x@W
  GEMM_CASE(2, 12)   // C=64:  GRU gates (3C=192)
  GEMM_CASE(4, 8)    // C=128: x@W
  GEMM_CASE(4, 24)   // C=128: GRU gates (3C=384)
  GEMM_CASE(4, 16)   // fc1: 128 -> 256
#undef GEMM_CASE
}

// ---------------------------------------------------------------------------
// fc2 + log_softmax: one block per graph, K=256 threads, A=10 classes
// ---------------------------------------------------------------------------
__global__ void k_fc2_lsm(const float* __restrict__ act, const float* __restrict__ W,
                          const float* __restrict__ b, float* __restrict__ out,
                          int K, int A) {
  int g = blockIdx.x, t = threadIdx.x;
  __shared__ float red[256];
  __shared__ float logits[16];
  float av = act[(size_t)g * K + t];
  for (int a = 0; a < A; ++a) {
    red[t] = av * W[(size_t)a * K + t];
    __syncthreads();
    for (int s = 128; s > 0; s >>= 1) {
      if (t < s) red[t] += red[t + s];
      __syncthreads();
    }
    if (t == 0) logits[a] = red[0] + b[a];
    __syncthreads();
  }
  if (t == 0) {
    float mx = logits[0];
    for (int a = 1; a < A; ++a) mx = fmaxf(mx, logits[a]);
    float se = 0.f;
    for (int a = 0; a < A; ++a) se += expf(logits[a] - mx);
    float lse = mx + logf(se);
    for (int a = 0; a < A; ++a) out[(size_t)g * A + a] = logits[a] - lse;
  }
}

// ---------------------------------------------------------------------------
// Orchestration
// ---------------------------------------------------------------------------
extern "C" void kernel_launch(void* const* d_in, const int* in_sizes, int n_in,
                              void* d_out, int out_size, void* d_ws, size_t ws_size,
                              hipStream_t stream) {
  const int NF = 32;
  const int N  = in_sizes[0] / NF;          // 50000
  const int E  = in_sizes[1] / 2;           // 800000
  const int G  = 256;                       // NUM_GRAPHS
  const int NA = in_sizes[6];               // 10

  const float*     x_in  = (const float*)d_in[0];
  const long long* ei    = (const long long*)d_in[1];
  const long long* batch = (const long long*)d_in[2];
  const float*     fc1_W = (const float*)d_in[3];
  const float*     fc1_b = (const float*)d_in[4];
  const float*     fc2_W = (const float*)d_in[5];
  const float*     fc2_b = (const float*)d_in[6];

  const float *convW[3], *wih[3], *whh[3], *bih[3], *bhh[3];
  for (int s = 0; s < 3; ++s) {
    convW[s] = (const float*)d_in[7 + 5 * s + 0];
    wih[s]   = (const float*)d_in[7 + 5 * s + 1];
    whh[s]   = (const float*)d_in[7 + 5 * s + 2];
    bih[s]   = (const float*)d_in[7 + 5 * s + 3];
    bhh[s]   = (const float*)d_in[7 + 5 * s + 4];
  }
  const float* bn_g[2] = {(const float*)d_in[22], (const float*)d_in[26]};
  const float* bn_b[2] = {(const float*)d_in[23], (const float*)d_in[27]};
  const float* bn_m[2] = {(const float*)d_in[24], (const float*)d_in[28]};
  const float* bn_v[2] = {(const float*)d_in[25], (const float*)d_in[29]};

  // ---- workspace carve (256B aligned); m aliases gi, agg aliases gh (lifetimes disjoint)
  char*  ws  = (char*)d_ws;
  size_t off = 0;
  auto carve = [&](size_t bytes) -> char* {
    char* p = ws + off;
    off += (bytes + 255) & ~(size_t)255;
    return p;
  };
  float*          xf[2];
  unsigned short* xb[2];
  xf[0] = (float*)carve((size_t)N * 128 * 4);
  xb[0] = (unsigned short*)carve((size_t)N * 128 * 2);
  xf[1] = (float*)carve((size_t)N * 128 * 4);
  xb[1] = (unsigned short*)carve((size_t)N * 128 * 2);
  float*          gi    = (float*)carve((size_t)N * 384 * 4);
  float*          gh    = (float*)carve((size_t)N * 384 * 4);
  unsigned short* aggb  = (unsigned short*)carve((size_t)N * 128 * 2);
  unsigned short* pw    = (unsigned short*)carve(128 * 128 * 2);
  unsigned short* pwih  = (unsigned short*)carve(128 * 384 * 2);
  unsigned short* pwhh  = (unsigned short*)carve(128 * 384 * 2);
  float*          pool  = (float*)carve((size_t)G * 128 * 4);
  unsigned short* poolb = (unsigned short*)carve((size_t)G * 128 * 2);
  unsigned short* pfc1  = (unsigned short*)carve(128 * 256 * 2);
  float*          fc1o  = (float*)carve((size_t)G * 256 * 4);
  float*          fc1a  = (float*)carve((size_t)G * 256 * 4);
  unsigned short* fc1ab = (unsigned short*)carve((size_t)G * 256 * 2);
  float* mf  = gi;   // [N,C] message buffer, dead before gi is produced
  float* agg = gh;   // [N,C] aggregate buffer, dead before gh is produced

  const int TB = 256;
  // stage 0 init: x (NF == C1 == 32, no pad)
  k_copy_convert4<<<ceil_div(N * 8, TB), TB, 0, stream>>>(
      (const float4*)x_in, (float4*)xf[0], xb[0], N * 8);

  const int Cs[3] = {32, 64, 128};
  for (int s = 0; s < 3; ++s) {
    const int C   = Cs[s];
    const int buf = s & 1;                       // stage 0 -> buf0, 1 -> buf1, 2 -> buf0
    float*          xfc = xf[buf];
    unsigned short* xbc = xb[buf];

    k_pack_B<<<ceil_div(C * 3 * C, TB), TB, 0, stream>>>(wih[s], pwih, C, 3 * C, 1);
    k_pack_B<<<ceil_div(C * 3 * C, TB), TB, 0, stream>>>(whh[s], pwhh, C, 3 * C, 1);

    for (int l = 0; l < 4; ++l) {
      k_pack_B<<<ceil_div(C * C, TB), TB, 0, stream>>>(convW[s] + (size_t)l * C * C, pw, C, C, 0);
      gemm(xbc, pw, nullptr, mf, N, C, C, stream);                         // m = x @ W[l]
      k_fill_zero4<<<ceil_div(N * (C >> 2), TB), TB, 0, stream>>>((float4*)agg, N * (C >> 2));
      k_scatter_add<<<ceil_div(E * (C >> 2), TB), TB, 0, stream>>>(mf, ei, agg, E, C);
      k_to_bf16_4<<<ceil_div(N * (C >> 2), TB), TB, 0, stream>>>((const float4*)agg, aggb, N * (C >> 2));
      gemm(aggb, pwih, bih[s], gi, N, C, 3 * C, stream);                   // gi = agg @ w_ih^T + b_ih
      gemm(xbc,  pwhh, bhh[s], gh, N, C, 3 * C, stream);                   // gh = x   @ w_hh^T + b_hh
      k_gru<<<ceil_div(N * (C >> 2), TB), TB, 0, stream>>>(gi, gh, xfc, xbc, N, C);
    }

    if (s == 0)
      k_transition<<<ceil_div(N * 16, TB), TB, 0, stream>>>(
          xfc, 32, xf[1], xb[1], 64, bn_g[0], bn_b[0], bn_m[0], bn_v[0], N);
    else if (s == 1)
      k_transition<<<ceil_div(N * 32, TB), TB, 0, stream>>>(
          xfc, 64, xf[0], xb[0], 128, bn_g[1], bn_b[1], bn_m[1], bn_v[1], N);
    else
      k_transition<<<ceil_div(N * 32, TB), TB, 0, stream>>>(
          xfc, 128, xf[1], xb[1], 128, nullptr, nullptr, nullptr, nullptr, N);
  }

  // global_add_pool over sorted batch -> [G,128]
  k_fill_zero4<<<ceil_div(G * 32, TB), TB, 0, stream>>>((float4*)pool, G * 32);
  k_pool<<<ceil_div(N * 32, TB), TB, 0, stream>>>(xf[1], batch, pool, N, 128);
  k_to_bf16_4<<<ceil_div(G * 32, TB), TB, 0, stream>>>((const float4*)pool, poolb, G * 32);

  // fc1 (WMMA, bias fused) + ELU, then fc2 + log_softmax
  k_pack_B<<<ceil_div(128 * 256, TB), TB, 0, stream>>>(fc1_W, pfc1, 128, 256, 1);
  gemm(poolb, pfc1, fc1_b, fc1o, G, 128, 256, stream);
  k_transition<<<ceil_div(G * 64, TB), TB, 0, stream>>>(
      fc1o, 256, fc1a, fc1ab, 256, nullptr, nullptr, nullptr, nullptr, G);
  k_fc2_lsm<<<G, 256, 0, stream>>>(fc1a, fc2_W, fc2_b, (float*)d_out, 256, NA);
}
